// RSNN_LSTM_29257317220491
// MI455X (gfx1250) — matrix operand order repaired and checked by reading, not compiled, not run on hardware
//
#include <hip/hip_runtime.h>
#include <hip/hip_bf16.h>

typedef __bf16 bf16;
typedef __attribute__((ext_vector_type(16))) __bf16 v16bf;
typedef __attribute__((ext_vector_type(8)))  __bf16 v8bf;
typedef __attribute__((ext_vector_type(8)))  float  v8f;

#define T_STEPS 2048
#define I_DIM   2048
#define H_DIM   8192
#define O_DIM   1024

// ---------------- conversion / prep kernels ----------------

__global__ void cvt_f32_bf16(const float* __restrict__ in, bf16* __restrict__ out, size_t n) {
  size_t i = (size_t)blockIdx.x * blockDim.x + threadIdx.x;
  size_t stride = (size_t)gridDim.x * blockDim.x;
  for (; i < n; i += stride) out[i] = (bf16)in[i];
}

__global__ void extract_diag(const float* __restrict__ Rw, float* __restrict__ dg, int n) {
  int i = blockIdx.x * blockDim.x + threadIdx.x;
  if (i < n) dg[i] = Rw[(size_t)i * n + i];
}

// ---------------- CDNA5 async staging helpers ----------------

// GLOBAL_LOAD_ASYNC_TO_LDS_B128, GVS mode: mem = SADDR(64b, uniform) + VADDR(32b) ;
// each active lane moves 16B into LDS[VDST]. Tracked by ASYNCcnt.
__device__ __forceinline__ void async_ld_b128(uint32_t lds_addr, const bf16* sbase, uint32_t voff) {
  asm volatile("global_load_async_to_lds_b128 %0, %1, %2"
               :: "v"(lds_addr), "v"(voff), "s"(sbase)
               : "memory");
}

__device__ __forceinline__ void wait_asynccnt0() {
  asm volatile("s_wait_asynccnt 0x0" ::: "memory");
}

// ---------------- WMMA GEMM: C[m,n] = act( sum_k A[m,k]*B[n,k] + bias[n] ) ----------------
// A: [M,K] bf16 row-major, B: [N,K] bf16 row-major (i.e. computes A @ B^T).
// Block tile 128x128, BK=32, 8 waves (4 along M x 2 along N), wave tile 32x64 (2x4 WMMA tiles).
// K-tiles staged into double-buffered LDS with global_load_async_to_lds_b128 (ASYNCcnt).

__device__ __forceinline__ v16bf cat16(v8bf lo, v8bf hi) {
  return __builtin_shufflevector(lo, hi, 0,1,2,3,4,5,6,7,8,9,10,11,12,13,14,15);
}

template <int DO_SIGMOID>
__global__ __launch_bounds__(256)
void wmma_gemm_nt(const bf16* __restrict__ A, const bf16* __restrict__ B,
                  const float* __restrict__ bias, float* __restrict__ C,
                  int M, int N, int K)
{
  constexpr int BM = 128, BN = 128, BK = 32, LDK = BK + 16; // pad keeps b128 alignment (96B rows)
  __shared__ bf16 As[2][BM][LDK];
  __shared__ bf16 Bs[2][BN][LDK];
  constexpr uint32_t BUFSZ = BM * LDK * sizeof(bf16);       // 12288 B per buffer

  const int tid   = threadIdx.x;
  const int lane  = tid & 31;
  const int wave  = tid >> 5;
  const int lm    = lane & 15;   // M index (A frag) / N index (B frag) within 16x16 tile
  const int half  = lane >> 4;   // lane half selects K sub-range per ISA layout
  const int waveM = (wave & 3) * 32;
  const int waveN = (wave >> 2) * 64;
  const int m0 = blockIdx.y * BM;
  const int n0 = blockIdx.x * BN;

  // cooperative staging: each thread moves 2x16B for A tile and 2x16B for B tile
  const int srow = tid >> 1;         // 0..127
  const int scol = (tid & 1) * 16;   // 0 or 16 (bf16 elems)

  const bf16* gAbase = A + (size_t)m0 * K;   // uniform SGPR base
  const bf16* gBbase = B + (size_t)n0 * K;   // uniform SGPR base
  const uint32_t voBase = (uint32_t)((srow * K + scol) * sizeof(bf16)); // per-lane byte offset
  const uint32_t ldsA0 = (uint32_t)(size_t)&As[0][srow][scol];          // flat addr[31:0] == DS addr
  const uint32_t ldsB0 = (uint32_t)(size_t)&Bs[0][srow][scol];

  v8f acc[2][4];
#pragma unroll
  for (int i = 0; i < 2; ++i)
#pragma unroll
    for (int j = 0; j < 4; ++j)
      acc[i][j] = (v8f){0.f,0.f,0.f,0.f,0.f,0.f,0.f,0.f};

  const int NK = K / BK;

  // prologue: async-stage K-tile 0 into buffer 0
  {
    const uint32_t vo = voBase;
    async_ld_b128(ldsA0,      gAbase, vo);
    async_ld_b128(ldsA0 + 16, gAbase, vo + 16);
    async_ld_b128(ldsB0,      gBbase, vo);
    async_ld_b128(ldsB0 + 16, gBbase, vo + 16);
  }
  wait_asynccnt0();
  __syncthreads();

  for (int kt = 0; kt < NK; ++kt) {
    const int cur = kt & 1;
    if (kt + 1 < NK) {   // async-stage next K-tile into the other buffer (overlaps WMMA below)
      const uint32_t nb = (uint32_t)(cur ^ 1) * BUFSZ;
      const uint32_t vo = voBase + (uint32_t)(kt + 1) * (BK * sizeof(bf16));
      async_ld_b128(ldsA0 + nb,      gAbase, vo);
      async_ld_b128(ldsA0 + nb + 16, gAbase, vo + 16);
      async_ld_b128(ldsB0 + nb,      gBbase, vo);
      async_ld_b128(ldsB0 + nb + 16, gBbase, vo + 16);
    }

    // A fragment: elems 0..7 <- K = half*8 + 0..7 ; elems 8..15 <- K = 16 + half*8 + 0..7
    v16bf afrag[2];
#pragma unroll
    for (int ti = 0; ti < 2; ++ti) {
      const bf16* p = &As[cur][waveM + ti * 16 + lm][0];
      afrag[ti] = cat16(*(const v8bf*)(p + half * 8),
                        *(const v8bf*)(p + 16 + half * 8));
    }
    // B fragment: lane = N, elems 0..7 <- K = half*8 + 0..7 ; elems 8..15 <- K = 16 + half*8 + 0..7
    v16bf bfrag[4];
#pragma unroll
    for (int tj = 0; tj < 4; ++tj) {
      const bf16* p = &Bs[cur][waveN + tj * 16 + lm][0];
      bfrag[tj] = cat16(*(const v8bf*)(p + half * 8),
                        *(const v8bf*)(p + 16 + half * 8));
    }

#pragma unroll
    for (int ti = 0; ti < 2; ++ti)
#pragma unroll
      for (int tj = 0; tj < 4; ++tj)
        acc[ti][tj] = __builtin_amdgcn_wmma_f32_16x16x32_bf16(
            false, afrag[ti], false, bfrag[tj], (short)0, acc[ti][tj], false, false);

    if (kt + 1 < NK) {
      wait_asynccnt0();   // my async stores into LDS landed
      __syncthreads();    // everyone's landed + everyone done reading `cur`
    }
  }

  // epilogue: C/D layout — lanes 0-15: N=lane, M=r ; lanes 16-31: N=lane-16, M=8+r
#pragma unroll
  for (int tj = 0; tj < 4; ++tj) {
    const int n = n0 + waveN + tj * 16 + lm;
    const float bn = bias[n];
#pragma unroll
    for (int ti = 0; ti < 2; ++ti) {
#pragma unroll
      for (int r = 0; r < 8; ++r) {
        const int m = m0 + waveM + ti * 16 + r + half * 8;
        float v = acc[ti][tj][r] + bn;
        if (DO_SIGMOID) v = 1.0f / (1.0f + __expf(-v));
        C[(size_t)m * N + n] = v;
      }
    }
  }
}

// ---------------- elementwise LIF scans (only truly sequential part) ----------------

// Layer 1: mem1_rec initially holds candidate = sigmoid(XW1^T+b1); overwritten in place.
__global__ void lif1_scan(const float* __restrict__ dg,
                          float* __restrict__ spk1_rec,
                          float* __restrict__ mem1_rec,
                          float* __restrict__ hid_rec,
                          bf16*  __restrict__ spk1bf)
{
  const int h = blockIdx.x * blockDim.x + threadIdx.x;
  if (h >= H_DIM) return;
  const float rw = dg[h];
  float mem = 0.f, syn = 0.f, hid = 0.f;   // hid = spk1 from previous step
  for (int t = 0; t < T_STEPS; ++t) {
    const size_t idx = (size_t)t * H_DIM + h;
    const float cand  = mem1_rec[idx];                 // read-before-write aliasing
    const float fgate = fmaxf(rw * hid, 0.f);          // relu(diag(Rw) * spk_prev)
    syn = fgate * syn + (1.f - fgate) * cand;
    const float spk = (mem - 1.0f) > 0.f ? 1.f : 0.f;  // spike from OLD mem
    mem = 0.5f * mem + syn - spk;                      // BETA=0.5, THR=1
    spk1_rec[idx] = spk;
    mem1_rec[idx] = mem;
    hid_rec[idx]  = hid;                               // carry value BEFORE update
    spk1bf[idx]   = (bf16)spk;                         // exact in bf16
    hid = spk;
  }
}

// Layer 2: mem2_rec initially holds i2 = spk1 @ W2^T + b2; overwritten in place.
__global__ void lif2_scan(float* __restrict__ spk2_rec, float* __restrict__ mem2_rec)
{
  const int o = blockIdx.x * blockDim.x + threadIdx.x;
  if (o >= O_DIM) return;
  float mem = 0.f, syn = 0.f;
  for (int t = 0; t < T_STEPS; ++t) {
    const size_t idx = (size_t)t * O_DIM + o;
    const float i2  = mem2_rec[idx];
    const float spk = (mem - 1.0f) > 0.f ? 1.f : 0.f;
    syn = 0.5f * syn + i2;
    mem = 0.5f * mem + syn - spk;
    spk2_rec[idx] = spk;
    mem2_rec[idx] = mem;
  }
}

// ---------------- launcher ----------------

extern "C" void kernel_launch(void* const* d_in, const int* in_sizes, int n_in,
                              void* d_out, int out_size, void* d_ws, size_t ws_size,
                              hipStream_t stream)
{
  (void)in_sizes; (void)n_in; (void)out_size; (void)ws_size;

  const float* x  = (const float*)d_in[0];   // [T,I]
  const float* W1 = (const float*)d_in[1];   // [H,I]
  const float* b1 = (const float*)d_in[2];   // [H]
  const float* Rw = (const float*)d_in[3];   // [H,H] (diagonal)
  const float* W2 = (const float*)d_in[4];   // [O,H]
  const float* b2 = (const float*)d_in[5];   // [O]

  float* out = (float*)d_out;
  const size_t TH = (size_t)T_STEPS * H_DIM;
  const size_t TO = (size_t)T_STEPS * O_DIM;
  float* spk1_rec = out;                     // [T,H]
  float* mem1_rec = out + TH;                // [T,H]  (temp: candidate)
  float* spk2_rec = out + 2 * TH;            // [T,O]
  float* mem2_rec = out + 2 * TH + TO;       // [T,O]  (temp: i2)
  float* hid_rec  = out + 2 * TH + 2 * TO;   // [T,H]

  // workspace carve (all 256B aligned)
  uint8_t* ws = (uint8_t*)d_ws;
  size_t off = 0;
  bf16*  Xbf  = (bf16*) (ws + off); off += (size_t)T_STEPS * I_DIM * 2;  // 8 MB
  bf16*  W1bf = (bf16*) (ws + off); off += (size_t)H_DIM  * I_DIM * 2;   // 32 MB
  bf16*  W2bf = (bf16*) (ws + off); off += (size_t)O_DIM  * H_DIM * 2;   // 16 MB
  float* dg   = (float*)(ws + off); off += (size_t)H_DIM * 4;            // 32 KB
  bf16*  S1bf = (bf16*) (ws + off);                                      // 32 MB

  // prep
  cvt_f32_bf16<<<2048, 256, 0, stream>>>(x,  Xbf,  (size_t)T_STEPS * I_DIM);
  cvt_f32_bf16<<<2048, 256, 0, stream>>>(W1, W1bf, (size_t)H_DIM  * I_DIM);
  cvt_f32_bf16<<<2048, 256, 0, stream>>>(W2, W2bf, (size_t)O_DIM  * H_DIM);
  extract_diag<<<H_DIM / 256, 256, 0, stream>>>(Rw, dg, H_DIM);

  // GEMM1: candidate = sigmoid(X @ W1^T + b1)  -> mem1_rec (temp)
  dim3 grid1(H_DIM / 128, T_STEPS / 128);
  wmma_gemm_nt<1><<<grid1, 256, 0, stream>>>(Xbf, W1bf, b1, mem1_rec,
                                             T_STEPS, H_DIM, I_DIM);

  // layer-1 LIF scan (produces spk1, mem1, hid records + bf16 spikes)
  lif1_scan<<<H_DIM / 256, 256, 0, stream>>>(dg, spk1_rec, mem1_rec, hid_rec, S1bf);

  // GEMM2: i2 = spk1 @ W2^T + b2 -> mem2_rec (temp)
  dim3 grid2(O_DIM / 128, T_STEPS / 128);
  wmma_gemm_nt<0><<<grid2, 256, 0, stream>>>(S1bf, W2bf, b2, mem2_rec,
                                             T_STEPS, O_DIM, H_DIM);

  // layer-2 LIF scan
  lif2_scan<<<O_DIM / 256, 256, 0, stream>>>(spk2_rec, mem2_rec);
}
